// MultiLevel_BlockAttention_15625091023263
// MI455X (gfx1250) — matrix operand order repaired and checked by reading, not compiled, not run on hardware
//
#include <hip/hip_runtime.h>
#include <hip/hip_bf16.h>
#include <math.h>

typedef __bf16 bf16;
typedef __attribute__((ext_vector_type(16))) __bf16 v16bf;
typedef __attribute__((ext_vector_type(8)))  __bf16 v8bf;
typedef __attribute__((ext_vector_type(4)))  __bf16 v4bf;
typedef __attribute__((ext_vector_type(8)))  float  v8f;

#define WMMA_BF16(a, b, c) \
  __builtin_amdgcn_wmma_f32_16x16x32_bf16(false, (a), false, (b), (short)0, (c), false, false)

constexpr int BATCH = 2;
constexpr int SEQ   = 2048;
constexpr int EMB   = 1024;
constexpr int NH    = 8;
constexpr int HD    = 128;           // head dim
constexpr int BS    = BATCH * SEQ;   // 4096
__device__ constexpr float SCALE = 0.022097086912079608f; // 1/sqrt(2048)

// ---------------------------------------------------------------------------
// Fragment helpers (wave32 WMMA, 16-bit operands; contiguous K order used by A and B here)
// A 16x32: lane l (l16 = l&15, half = l>>4): row M=l16,
//   16 contiguous K starting at half*16 (matches B)
// B 32x16: lane holds column N=l16, 16 contiguous K starting at half*16
// C/D f32 16x16: elem r of v8f: (M = r + 8*half, N = l16)
// ---------------------------------------------------------------------------
__device__ inline v16bf frag_a(const bf16* __restrict__ row, int half) {
  v8bf lo = *(const v8bf*)(row + half * 16);
  v8bf hi = *(const v8bf*)(row + half * 16 + 8);
  v16bf r;
#pragma unroll
  for (int i = 0; i < 8; ++i) { r[i] = lo[i]; r[i + 8] = hi[i]; }
  return r;
}

__device__ inline v16bf frag_b16(const bf16* __restrict__ p) {
  v8bf lo = *(const v8bf*)p;
  v8bf hi = *(const v8bf*)(p + 8);
  v16bf r;
#pragma unroll
  for (int i = 0; i < 8; ++i) { r[i] = lo[i]; r[i + 8] = hi[i]; }
  return r;
}

__device__ inline float redmax16(float v) {
#pragma unroll
  for (int off = 8; off; off >>= 1) v = fmaxf(v, __shfl_xor(v, off, 32));
  return v;
}
__device__ inline float redsum16(float v) {
#pragma unroll
  for (int off = 8; off; off >>= 1) v += __shfl_xor(v, off, 32);
  return v;
}

// ---------------------------------------------------------------------------
// Stage 1a: f32 -> bf16 conversion (vectorized 4-wide)
// ---------------------------------------------------------------------------
__global__ void cvt_kernel(const float* __restrict__ in, bf16* __restrict__ out, int n4) {
  int i = blockIdx.x * blockDim.x + threadIdx.x;
  int stride = gridDim.x * blockDim.x;
  for (; i < n4; i += stride) {
    float4 f = ((const float4*)in)[i];
    v4bf o;
    o[0] = (bf16)f.x; o[1] = (bf16)f.y; o[2] = (bf16)f.z; o[3] = (bf16)f.w;
    ((v4bf*)out)[i] = o;
  }
}

// ---------------------------------------------------------------------------
// Stage 1b: f32 W[k][n] -> bf16 Wt[n][k] (tiled transpose via LDS).
// ---------------------------------------------------------------------------
__global__ __launch_bounds__(256) void cvt_transpose_kernel(
    const float* __restrict__ in, bf16* __restrict__ out) {
  __shared__ float tile[32][33];
  const int k0 = blockIdx.x * 32, n0 = blockIdx.y * 32;
  const int tx = threadIdx.x, ty = threadIdx.y;  // 32 x 8
#pragma unroll
  for (int i = 0; i < 32; i += 8)
    tile[ty + i][tx] = in[(size_t)(k0 + ty + i) * EMB + n0 + tx];
  __syncthreads();
#pragma unroll
  for (int i = 0; i < 32; i += 8)
    out[(size_t)(n0 + ty + i) * EMB + k0 + tx] = (bf16)tile[tx][ty + i];
}

// ---------------------------------------------------------------------------
// Stage 2: QKV projection GEMM, weights pre-transposed (Wt[n][k]).
// Block tile 128x128, 8 waves (4x2), wave tile 32x64, K-step 32.
// ---------------------------------------------------------------------------
__global__ __launch_bounds__(256) void qkv_gemm(
    const bf16* __restrict__ Xb,
    const bf16* __restrict__ Wqt, const bf16* __restrict__ Wkt, const bf16* __restrict__ Wvt,
    const float* __restrict__ bq, const float* __restrict__ bk, const float* __restrict__ bv,
    bf16* __restrict__ qo, bf16* __restrict__ ko, bf16* __restrict__ vo) {
  const bf16* Wt;  const float* bias;  bf16* out;
  if (blockIdx.z == 0)      { Wt = Wqt; bias = bq; out = qo; }
  else if (blockIdx.z == 1) { Wt = Wkt; bias = bk; out = ko; }
  else                      { Wt = Wvt; bias = bv; out = vo; }

  __shared__ bf16 As[128 * 40];
  __shared__ bf16 Bs[128 * 40];

  const int tid  = threadIdx.x;
  const int wave = tid >> 5;
  const int lane = tid & 31;
  const int half = lane >> 4;
  const int l16  = lane & 15;
  const int waveM = wave >> 1;
  const int waveN = wave & 1;

  const int row0 = blockIdx.x * 128;
  const int col0 = blockIdx.y * 128;

  const int lmA  = tid >> 3;
  const int lk4A = (tid & 7) * 4;

  v8f acc[2][4];
#pragma unroll
  for (int mt = 0; mt < 2; ++mt)
#pragma unroll
    for (int nt = 0; nt < 4; ++nt)
#pragma unroll
      for (int i = 0; i < 8; ++i) acc[mt][nt][i] = 0.f;

  for (int k0 = 0; k0 < EMB; k0 += 32) {
    __syncthreads();
#pragma unroll
    for (int rep = 0; rep < 4; ++rep) {
      int m = rep * 32 + lmA;
      v4bf a4 = *(const v4bf*)(Xb + (size_t)(row0 + m) * EMB + k0 + lk4A);
      *(v4bf*)(As + m * 40 + lk4A) = a4;
      v4bf w4 = *(const v4bf*)(Wt + (size_t)(col0 + m) * EMB + k0 + lk4A);
      *(v4bf*)(Bs + m * 40 + lk4A) = w4;
    }
    __syncthreads();

    v16bf af[2];
#pragma unroll
    for (int mt = 0; mt < 2; ++mt)
      af[mt] = frag_a(As + (waveM * 32 + mt * 16 + l16) * 40, half);
    v16bf bfr[4];
#pragma unroll
    for (int nt = 0; nt < 4; ++nt)
      bfr[nt] = frag_b16(Bs + (waveN * 64 + nt * 16 + l16) * 40 + half * 16);

#pragma unroll
    for (int mt = 0; mt < 2; ++mt)
#pragma unroll
      for (int nt = 0; nt < 4; ++nt)
        acc[mt][nt] = WMMA_BF16(af[mt], bfr[nt], acc[mt][nt]);
  }

#pragma unroll
  for (int mt = 0; mt < 2; ++mt) {
#pragma unroll
    for (int nt = 0; nt < 4; ++nt) {
#pragma unroll
      for (int r = 0; r < 8; ++r) {
        int m = row0 + waveM * 32 + mt * 16 + r + 8 * half;
        int n = col0 + waveN * 64 + nt * 16 + l16;
        float v = acc[mt][nt][r] + bias[m & (SEQ - 1)];
        out[(size_t)m * EMB + n] = (bf16)v;
      }
    }
  }
}

// ---------------------------------------------------------------------------
// Stage 3: flash attention, software-pipelined.
// K tile: triple-buffered LDS, filled by async global->LDS (ASYNCcnt); the
// prefetch for tile i+1 is issued before tile i's barrier and left in flight
// (s_wait_asynccnt 4). V tile: double-buffered LDS + register staging with a
// 4x4 register transpose. One barrier per iteration (skew < 1 iter, buffers
// chosen so a leading wave never touches the trailing wave's live buffer).
// ---------------------------------------------------------------------------
constexpr int KSZ = 32 * 136;   // one K buffer, [t][d] pad 136
constexpr int VSZ = 128 * 40;   // one V buffer, [d][t] pad 40

__global__ __launch_bounds__(128) void attn_kernel(
    const bf16* __restrict__ qb, const bf16* __restrict__ kb, const bf16* __restrict__ vb,
    const float* __restrict__ residual, const float* __restrict__ Wepi,
    const float* __restrict__ bo,
    float* __restrict__ effect_out, float* __restrict__ attn_out) {
  __shared__ bf16 Ks[3 * KSZ];         // triple-buffered K
  __shared__ bf16 Vt[2 * VSZ];         // double-buffered V (transposed)
  __shared__ bf16 Ps[4 * 32 * 40];     // per-wave P staging

  const int sblk = blockIdx.x;
  const int bh   = blockIdx.y;
  const int b    = bh >> 3;
  const int h    = bh & 7;

  const int tid  = threadIdx.x;
  const int wave = tid >> 5;           // 0..3
  const int lane = tid & 31;
  const int half = lane >> 4;
  const int l16  = lane & 15;

  const int m0 = sblk * 128 + wave * 32;

  // preload Q fragments: 2 M-subtiles x 4 K-steps
  v16bf qf[2][4];
#pragma unroll
  for (int mt = 0; mt < 2; ++mt) {
    const size_t qrow = ((size_t)b * SEQ + m0 + mt * 16 + l16) * EMB + (size_t)h * HD;
#pragma unroll
    for (int kq = 0; kq < 4; ++kq)
      qf[mt][kq] = frag_a(qb + qrow + kq * 32, half);
  }

  v8f O[2][8];
#pragma unroll
  for (int mt = 0; mt < 2; ++mt)
#pragma unroll
    for (int nd = 0; nd < 8; ++nd)
#pragma unroll
      for (int i = 0; i < 8; ++i) O[mt][nd][i] = 0.f;
  float mrow[2][8], lrow[2][8];
#pragma unroll
  for (int mt = 0; mt < 2; ++mt)
#pragma unroll
    for (int r = 0; r < 8; ++r) { mrow[mt][r] = -INFINITY; lrow[mt][r] = 0.f; }

  const unsigned ks_lds0 = (unsigned)(size_t)(void*)Ks;  // LDS offset = low 32 flat bits
  const int d8 = l16 * 8;

  // async K-tile issue: 4 ops/wave, each lane moves 16 B
  auto issueK = [&](int t0, int buf) {
#pragma unroll
    for (int rep = 0; rep < 4; ++rep) {
      int t = rep * 8 + wave * 2 + half;               // 0..31 disjoint over block
      unsigned ldsoff = ks_lds0 + (unsigned)((buf * KSZ + t * 136 + d8) * 2);
      unsigned goff   = (unsigned)((((size_t)b * SEQ + t0 + t) * EMB
                                    + (size_t)h * HD + d8) * sizeof(bf16));
      asm volatile("global_load_async_to_lds_b128 %0, %1, %2"
                   :: "v"(ldsoff), "v"(goff), "s"(kb) : "memory");
    }
  };

  v4bf vreg[2][4];                                     // staged V rows
  auto loadV = [&](int t0) {
#pragma unroll
    for (int rep = 0; rep < 2; ++rep) {
      int tq = rep * 16 + wave * 4;
#pragma unroll
      for (int j = 0; j < 4; ++j)
        vreg[rep][j] = *(const v4bf*)(vb + ((size_t)b * SEQ + t0 + tq + j) * EMB
                                         + (size_t)h * HD + lane * 4);
    }
  };
  auto storeV = [&](bf16* vtb) {                       // 4x4 register transpose
#pragma unroll
    for (int rep = 0; rep < 2; ++rep) {
      int tq = rep * 16 + wave * 4;
#pragma unroll
      for (int i = 0; i < 4; ++i) {
        v4bf o;
#pragma unroll
        for (int j = 0; j < 4; ++j) o[j] = vreg[rep][j][i];
        *(v4bf*)(vtb + (lane * 4 + i) * 40 + tq) = o;
      }
    }
  };

  // pipeline prologue
  issueK(0, 0);
  loadV(0);

  for (int it = 0; it < SEQ / 32; ++it) {
    const int t0 = it * 32;
    bf16* ksc = Ks + (it % 3) * KSZ;
    bf16* vtc = Vt + (it & 1) * VSZ;

    storeV(vtc);                                      // commit staged V(it)
    if (it + 1 < SEQ / 32) {
      issueK(t0 + 32, (it + 1) % 3);                  // prefetch next K
      loadV(t0 + 32);                                 // prefetch next V into regs
      asm volatile("s_wait_asynccnt 0x4" ::: "memory");  // K(it) done, K(it+1) in flight
    } else {
      asm volatile("s_wait_asynccnt 0x0" ::: "memory");
    }
    __syncthreads();                                  // single barrier per iteration

    // --- S = q @ k^T : 2 M-subtiles x 2 N-subtiles, K frags shared ---
    v8f sc[2][2];
#pragma unroll
    for (int mt = 0; mt < 2; ++mt)
#pragma unroll
      for (int nt = 0; nt < 2; ++nt)
#pragma unroll
        for (int i = 0; i < 8; ++i) sc[mt][nt][i] = 0.f;
#pragma unroll
    for (int kq = 0; kq < 4; ++kq) {
      v16bf b0 = frag_b16(ksc + l16 * 136 + kq * 32 + half * 16);
      v16bf b1 = frag_b16(ksc + (16 + l16) * 136 + kq * 32 + half * 16);
#pragma unroll
      for (int mt = 0; mt < 2; ++mt) {
        sc[mt][0] = WMMA_BF16(qf[mt][kq], b0, sc[mt][0]);
        sc[mt][1] = WMMA_BF16(qf[mt][kq], b1, sc[mt][1]);
      }
    }

    // --- scale + residual, emit attn (NT), online softmax, P staging ---
    bf16* pwave = Ps + wave * (32 * 40);
#pragma unroll
    for (int mt = 0; mt < 2; ++mt) {
      float alpha[8];
#pragma unroll
      for (int r = 0; r < 8; ++r) {
        int srow = m0 + mt * 16 + r + 8 * half;
        size_t rbase = (size_t)srow * SEQ + t0;
        float s0 = sc[mt][0][r] * SCALE + residual[rbase + l16];
        float s1 = sc[mt][1][r] * SCALE + residual[rbase + 16 + l16];
        size_t abase = ((size_t)bh * SEQ + srow) * SEQ + t0;
        __builtin_nontemporal_store(s0, attn_out + abase + l16);
        __builtin_nontemporal_store(s1, attn_out + abase + 16 + l16);

        float mx = redmax16(fmaxf(s0, s1));
        float mnew = fmaxf(mrow[mt][r], mx);
        float a  = __expf(mrow[mt][r] - mnew);
        float p0 = __expf(s0 - mnew);
        float p1 = __expf(s1 - mnew);
        lrow[mt][r] = lrow[mt][r] * a + redsum16(p0 + p1);
        mrow[mt][r] = mnew;
        alpha[r] = a;

        float w0 = Wepi[rbase + l16];
        float w1 = Wepi[rbase + 16 + l16];
        bf16* prow = pwave + (mt * 16 + r + 8 * half) * 40;
        prow[l16]      = (bf16)(p0 * w0);
        prow[16 + l16] = (bf16)(p1 * w1);
      }
#pragma unroll
      for (int nd = 0; nd < 8; ++nd)
#pragma unroll
        for (int r = 0; r < 8; ++r) O[mt][nd][r] *= alpha[r];
    }

    // --- O += P @ V (V frags shared across both M-subtiles) ---
    v16bf pf[2];
#pragma unroll
    for (int mt = 0; mt < 2; ++mt)
      pf[mt] = frag_a(pwave + (mt * 16 + l16) * 40, half);
#pragma unroll
    for (int nd = 0; nd < 8; ++nd) {
      v16bf vf = frag_b16(vtc + (nd * 16 + l16) * 40 + half * 16);
#pragma unroll
      for (int mt = 0; mt < 2; ++mt)
        O[mt][nd] = WMMA_BF16(pf[mt], vf, O[mt][nd]);
    }
  }

  // --- finalize: O / l + bo, merged-head layout, NT stores ---
#pragma unroll
  for (int mt = 0; mt < 2; ++mt) {
    float invl[8];
#pragma unroll
    for (int r = 0; r < 8; ++r) invl[r] = 1.f / lrow[mt][r];
#pragma unroll
    for (int nd = 0; nd < 8; ++nd) {
#pragma unroll
      for (int r = 0; r < 8; ++r) {
        int srow = m0 + mt * 16 + r + 8 * half;
        int d = nd * 16 + l16;
        float v = O[mt][nd][r] * invl[r] + bo[srow];
        __builtin_nontemporal_store(
            v, effect_out + ((size_t)b * SEQ + srow) * EMB + (size_t)h * HD + d);
      }
    }
  }
}

// ---------------------------------------------------------------------------
// Launch
// ---------------------------------------------------------------------------
extern "C" void kernel_launch(void* const* d_in, const int* in_sizes, int n_in,
                              void* d_out, int out_size, void* d_ws, size_t ws_size,
                              hipStream_t stream) {
  const float* X        = (const float*)d_in[0];
  const float* residual = (const float*)d_in[1];
  const float* Wq       = (const float*)d_in[2];
  const float* Wk       = (const float*)d_in[3];
  const float* Wv       = (const float*)d_in[4];
  const float* Wepi     = (const float*)d_in[5];
  const float* bq       = (const float*)d_in[6];
  const float* bk       = (const float*)d_in[7];
  const float* bv       = (const float*)d_in[8];
  const float* bo       = (const float*)d_in[9];

  float* effect = (float*)d_out;
  float* attn   = (float*)d_out + (size_t)BATCH * SEQ * EMB;

  char* w = (char*)d_ws;
  bf16* Xb   = (bf16*)(w);                       // 8 MB
  bf16* Wqt  = (bf16*)(w + (8u << 20));          // 2 MB (transposed)
  bf16* Wkt  = (bf16*)(w + (10u << 20));
  bf16* Wvt  = (bf16*)(w + (12u << 20));
  bf16* qbuf = (bf16*)(w + (14u << 20));         // 8 MB each
  bf16* kbuf = (bf16*)(w + (22u << 20));
  bf16* vbuf = (bf16*)(w + (30u << 20));         // total 38 MB

  cvt_kernel<<<512, 256, 0, stream>>>(X, Xb, BS * EMB / 4);
  cvt_transpose_kernel<<<dim3(EMB / 32, EMB / 32), dim3(32, 8), 0, stream>>>(Wq, Wqt);
  cvt_transpose_kernel<<<dim3(EMB / 32, EMB / 32), dim3(32, 8), 0, stream>>>(Wk, Wkt);
  cvt_transpose_kernel<<<dim3(EMB / 32, EMB / 32), dim3(32, 8), 0, stream>>>(Wv, Wvt);

  qkv_gemm<<<dim3(BS / 128, EMB / 128, 3), 256, 0, stream>>>(
      Xb, Wqt, Wkt, Wvt, bq, bk, bv, qbuf, kbuf, vbuf);

  attn_kernel<<<dim3(SEQ / 128, BATCH * NH), 128, 0, stream>>>(
      qbuf, kbuf, vbuf, residual, Wepi, bo, effect, attn);
}
